// LocalAttention_83451214561781
// MI455X (gfx1250) — matrix operand order repaired and checked by reading, not compiled, not run on hardware
//
#include <hip/hip_runtime.h>

// ---------------------------------------------------------------------------
// Local windowed attention, fused pipeline for gfx1250 (MI455X):
//   pass 1: depthwise causal conv (128 taps, packed f32 channel-pairs) + RoPE
//           (+ q scale) -> f16 scratch
//   pass 2: windowed attention (128 q x 256 k) with v_wmma_f32_16x16x32_f16,
//           flash-style online softmax in transposed (S^T) layout; K tile
//           staged with global_load_async_to_lds_b128 (ASYNCcnt).
// ---------------------------------------------------------------------------

#define BB   32
#define NN   4096
#define DD   64
#define WSZ  128
#define WW   (NN / WSZ)          // 32 windows
#define SCALE_Q 0.125f           // 64^-0.5
#define MASKV  (-10000.0f)

typedef __attribute__((ext_vector_type(16))) _Float16 v16h;
typedef __attribute__((ext_vector_type(8)))  float    v8f;
typedef __attribute__((ext_vector_type(2)))  float    v2f;

struct FragH {                    // one WMMA f16 A/B operand (16 halves/lane)
    union { unsigned u[8]; v16h h; };
};

static __device__ inline v8f wmma_f16(v16h a, v16h b, v8f c) {
    // D = A(16x32 f16) * B(32x16 f16) + C(16x16 f32)
    return __builtin_amdgcn_wmma_f32_16x16x32_f16(
        /*neg_a=*/false, a, /*neg_b=*/false, b,
        /*c_mod=*/(short)0, c, /*reuse_a=*/false, /*reuse_b=*/false);
}

static __device__ inline unsigned pack2h(float a, float b) {
    union { _Float16 h[2]; unsigned u; } t;
    t.h[0] = (_Float16)a; t.h[1] = (_Float16)b;
    return t.u;
}

// ---------------------------------------------------------------------------
// Pass 1: depthwise causal conv + RoPE.  Grid (W, B), 256 threads.
// Thread = (channel PAIR 0..31, group of 16 consecutive output positions).
// float2 register ring buffer -> v_pk_fma_f32 (2x VALU rate), RoPE is
// thread-local (both channels of the rotation pair in one thread), and the
// f16 output store is one packed b32 per position (contiguous 128B/wave).
// ---------------------------------------------------------------------------
__global__ __launch_bounds__(256)
void conv_rope_f16_kernel(const float* __restrict__ q,  const float* __restrict__ k,
                          const float* __restrict__ v,  const float* __restrict__ wq,
                          const float* __restrict__ wk, const float* __restrict__ wv,
                          _Float16* __restrict__ hq, _Float16* __restrict__ hk,
                          _Float16* __restrict__ hv)
{
    const int w    = blockIdx.x;
    const int b    = blockIdx.y;
    const int n0   = w * WSZ;
    const int tid  = threadIdx.x;
    const int pair = tid & 31;     // channel pair: owns d = 2*pair, 2*pair+1
    const int grp  = tid >> 5;     // 0..7 position group
    const int i0   = grp * 16;     // first of 16 output positions

    // stride 33 v2f (=66 dwords) -> per-lane b64 reads cover all 64 banks
    __shared__ v2f xs2[256][33];   // input tile [pos n0-128 .. n0+127][pair]
    __shared__ v2f ws2[128][33];   // weights transposed [tap][pair]

    const float*   xin[3]  = {q, k, v};
    const float*   wtab[3] = {wq, wk, wv};
    _Float16*      otab[3] = {hq, hk, hv};

    for (int tsr = 0; tsr < 3; ++tsr) {
        const float* x    = xin[tsr];
        const float* wsrc = wtab[tsr];

        // ---- stage weights (coalesced along taps) and input tile ----
        for (int e = tid; e < 32 * 128; e += 256) {
            const int pr = e >> 7, t = e & 127;
            v2f wv2;
            wv2.x = wsrc[(2 * pr    ) * 128 + t];
            wv2.y = wsrc[(2 * pr + 1) * 128 + t];
            ws2[t][pr] = wv2;
        }
        for (int r = grp; r < 256; r += 8) {
            const int pos = n0 - 128 + r;
            v2f val; val.x = 0.0f; val.y = 0.0f;
            if (pos >= 0)
                val = *(const v2f*)(x + ((size_t)(b * NN + pos)) * DD + 2 * pair);
            xs2[r][pair] = val;
        }
        __syncthreads();

        // ---- conv: 16 consecutive outputs, sliding packed register ring ----
        v2f acc[16];
        v2f X[16];
        #pragma unroll
        for (int p = 0; p < 16; ++p) { acc[p].x = 0.0f; acc[p].y = 0.0f; }
        #pragma unroll
        for (int j = 0; j < 16; ++j) X[j] = xs2[i0 + 1 + j][pair];

        for (int tt = 0; tt < 8; ++tt) {
            #pragma unroll
            for (int u = 0; u < 16; ++u) {
                const int t = tt * 16 + u;
                const v2f wv2 = ws2[t][pair];
                #pragma unroll
                for (int p = 0; p < 16; ++p)
                    acc[p] = __builtin_elementwise_fma(X[(u + p) & 15], wv2, acc[p]);
                if (t < 127) X[u] = xs2[i0 + 17 + t][pair];
            }
        }

        // ---- RoPE (q,k) thread-local + optional q-scale, packed f16 store ----
        _Float16* outp = otab[tsr];
        if (tsr < 2) {
            const float theta = powf(10000.0f, -(float)(2 * pair) * (1.0f / 64.0f));
            const float sc    = (tsr == 0) ? SCALE_Q : 1.0f;
            #pragma unroll
            for (int p = 0; p < 16; ++p) {
                const int pos = n0 + i0 + p;
                float sv, cv;
                sincosf((float)pos * theta, &sv, &cv);
                const float r0 = acc[p].x * cv - acc[p].y * sv;
                const float r1 = acc[p].y * cv + acc[p].x * sv;
                *(unsigned*)(outp + ((size_t)(b * NN + pos)) * DD + 2 * pair) =
                    pack2h(r0 * sc, r1 * sc);
            }
        } else {
            #pragma unroll
            for (int p = 0; p < 16; ++p) {
                const int pos = n0 + i0 + p;
                *(unsigned*)(outp + ((size_t)(b * NN + pos)) * DD + 2 * pair) =
                    pack2h(acc[p].x, acc[p].y);
            }
        }
        __syncthreads();
    }
}

// ---------------------------------------------------------------------------
// Pass 2: local attention.  Grid (W, B), 256 threads = 8 waves.
// Wave owns one 16-query block.  S^T = K * Q^T via WMMA so each lane holds a
// single query column -> online softmax needs one shfl_xor(16) per chunk and
// per-lane scalar accumulator rescale (flash attention).  out^T = V^T * P^T.
// K tile is staged via async global->LDS DMA (ASYNCcnt path).
// ---------------------------------------------------------------------------
__global__ __launch_bounds__(256)
void local_attn_wmma_kernel(const _Float16* __restrict__ qh,
                            const _Float16* __restrict__ kh,
                            const _Float16* __restrict__ vh,
                            float* __restrict__ out)
{
    const int w     = blockIdx.x;
    const int b     = blockIdx.y;
    const int tid   = threadIdx.x;
    const int lane  = tid & 31;
    const int wvid  = tid >> 5;                 // 0..7 query block
    const int mrow  = lane & 15;                // fragment row / query-in-block
    const int off8  = (lane < 16) ? 0 : 8;      // C-fragment M offset

    // padded to keep 16-lane b128 fragment reads on disjoint bank quads
    __shared__ __align__(16) _Float16 Ks [256][72];   // [key][d]
    __shared__ __align__(16) _Float16 VTs[ 64][264];  // [d][key]

    // ---- stage K rows: async global->LDS b128 (per-lane dsaddr = low 32
    //      bits of the flat &Ks[row][c8]); zero-pad window staged with DS ----
    if (w == 0) {
        for (int e = tid; e < 1024; e += 256) {           // rows 0..127 = pad
            const int row = e >> 3, c8 = (e & 7) << 3;
            uint4 z; z.x = z.y = z.z = z.w = 0u;
            *(uint4*)&Ks[row][c8] = z;
        }
        for (int e = tid; e < 1024; e += 256) {           // rows 128..255 real
            const int row = 128 + (e >> 3), c8 = (e & 7) << 3;
            const int kpos = row - 128;
            const unsigned ldsoff = (unsigned)(uintptr_t)&Ks[row][c8];
            const unsigned goff   =
                (unsigned)((((size_t)(b * NN + kpos)) * DD + c8) * sizeof(_Float16));
            asm volatile("global_load_async_to_lds_b128 %0, %1, %2"
                         :: "v"(ldsoff), "v"(goff), "s"(kh) : "memory");
        }
    } else {
        for (int e = tid; e < 2048; e += 256) {
            const int row = e >> 3, c8 = (e & 7) << 3;
            const int kpos = (row < 128) ? ((w - 1) * WSZ + row)
                                         : (w * WSZ + row - 128);
            const unsigned ldsoff = (unsigned)(uintptr_t)&Ks[row][c8];
            const unsigned goff   =
                (unsigned)((((size_t)(b * NN + kpos)) * DD + c8) * sizeof(_Float16));
            asm volatile("global_load_async_to_lds_b128 %0, %1, %2"
                         :: "v"(ldsoff), "v"(goff), "s"(kh) : "memory");
        }
    }
    // ---- stage V transposed: read half2 pairs, scatter rows of VT ----
    for (int e = tid; e < 8192; e += 256) {
        const int row = e >> 5, dp = e & 31;
        unsigned bits = 0u;
        if (!(row < 128 && w == 0)) {
            const int kpos = (row < 128) ? ((w - 1) * WSZ + row)
                                         : (w * WSZ + row - 128);
            bits = *(const unsigned*)(vh + ((size_t)(b * NN + kpos)) * DD + 2 * dp);
        }
        union { unsigned u; _Float16 h[2]; } cv; cv.u = bits;
        VTs[2 * dp    ][row] = cv.h[0];
        VTs[2 * dp + 1][row] = cv.h[1];
    }

    // ---- Q^T B-fragments straight from global (16 contiguous halves) ----
    const int qloc = wvid * 16 + mrow;
    const int qpos = w * WSZ + qloc;
    FragH Bq[2];
    #pragma unroll
    for (int c = 0; c < 2; ++c) {
        const int d0 = c * 32 + ((lane < 16) ? 0 : 16);
        const _Float16* base = qh + ((size_t)(b * NN + qpos)) * DD + d0;
        ((uint4*)Bq[c].u)[0] = *(const uint4*)base;
        ((uint4*)Bq[c].u)[1] = *(const uint4*)(base + 8);
    }
    asm volatile("s_wait_asynccnt 0x0" ::: "memory");   // async K staging done
    __syncthreads();

    float m = -3.0e38f, l = 0.0f;
    v8f acc[4];
    const v8f zero8 = {0.f, 0.f, 0.f, 0.f, 0.f, 0.f, 0.f, 0.f};
    #pragma unroll
    for (int dt = 0; dt < 4; ++dt) acc[dt] = zero8;

    for (int chunk = 0; chunk < 8; ++chunk) {       // 32 keys per chunk
        // ---- S^T tiles: rows = keys, cols = queries ----
        v8f S0 = zero8, S1 = zero8;
        #pragma unroll
        for (int c = 0; c < 2; ++c) {               // d = 0..31, 32..63
            FragH a0, a1;
            const _Float16* r0 = &Ks[chunk * 32 +      mrow][c * 32 + 2 * off8];
            const _Float16* r1 = &Ks[chunk * 32 + 16 + mrow][c * 32 + 2 * off8];
            ((uint4*)a0.u)[0] = *(const uint4*)r0;
            ((uint4*)a0.u)[1] = *(const uint4*)(r0 + 8);
            ((uint4*)a1.u)[0] = *(const uint4*)r1;
            ((uint4*)a1.u)[1] = *(const uint4*)(r1 + 8);
            S0 = wmma_f16(a0.h, Bq[c].h, S0);
            S1 = wmma_f16(a1.h, Bq[c].h, S1);
        }

        // ---- masks (exact reference semantics) + online softmax ----
        float pv[16];
        float cmax = -3.0e38f;
        const bool firstHalf = (chunk < 4);
        #pragma unroll
        for (int t = 0; t < 2; ++t) {
            #pragma unroll
            for (int j = 0; j < 8; ++j) {
                float s = t ? S1[j] : S0[j];
                const int kl = chunk * 32 + t * 16 + off8 + j;
                bool masked;
                if (firstHalf) {
                    const int kp = (w - 1) * WSZ + kl;
                    masked = (w == 0) || (kp == 0) || (qpos < kp);
                } else {
                    const int kp = w * WSZ + kl - 128;
                    masked = (kp == 0) || (qpos < kp);
                }
                s = masked ? MASKV : s;
                pv[t * 8 + j] = s;
                cmax = fmaxf(cmax, s);
            }
        }
        cmax = fmaxf(cmax, __shfl_xor(cmax, 16, 32));   // lane pair = same query
        const float mnew = fmaxf(m, cmax);
        const float sc   = __expf(m - mnew);
        m = mnew;
        float ps = 0.0f;
        #pragma unroll
        for (int x = 0; x < 16; ++x) { pv[x] = __expf(pv[x] - mnew); ps += pv[x]; }
        ps += __shfl_xor(ps, 16, 32);
        l = l * sc + ps;
        #pragma unroll
        for (int dt = 0; dt < 4; ++dt) acc[dt] *= sc;   // per-lane rescale

        // ---- rebuild P^T B-fragment from C layout (one lane-pair swap) ----
        unsigned pk0[4], pk1[4], op0[4], op1[4];
        #pragma unroll
        for (int j = 0; j < 4; ++j) {
            pk0[j] = pack2h(pv[2 * j],     pv[2 * j + 1]);      // tile0: keys off8+0..7
            pk1[j] = pack2h(pv[8 + 2 * j], pv[9 + 2 * j]);      // tile1
        }
        #pragma unroll
        for (int j = 0; j < 4; ++j) {
            op0[j] = (unsigned)__shfl_xor((int)pk0[j], 16, 32);
            op1[j] = (unsigned)__shfl_xor((int)pk1[j], 16, 32);
        }
        FragH Bp;
        #pragma unroll
        for (int j = 0; j < 4; ++j) {
            Bp.u[j]     = (lane < 16) ? pk0[j] : op1[j];  // keys 0..7  / 16..23
            Bp.u[4 + j] = (lane < 16) ? op0[j] : pk1[j];  // keys 8..15 / 24..31
        }

        // ---- out^T += V^T(chunk) * P^T(chunk) ----
        const int key0 = chunk * 32;
        #pragma unroll
        for (int dt = 0; dt < 4; ++dt) {
            FragH a;
            const _Float16* r = &VTs[dt * 16 + mrow][key0 + 2 * off8];
            ((uint4*)a.u)[0] = *(const uint4*)r;
            ((uint4*)a.u)[1] = *(const uint4*)(r + 8);
            acc[dt] = wmma_f16(a.h, Bp.h, acc[dt]);
        }
    }

    // ---- normalize and store out[b, qpos, :] (two float4 per d-tile) ----
    const float inv = 1.0f / l;
    #pragma unroll
    for (int dt = 0; dt < 4; ++dt) {
        const int dbase = dt * 16 + off8;
        float* o = out + ((size_t)(b * NN + qpos)) * DD + dbase;
        float4 lo, hi;
        lo.x = acc[dt][0] * inv; lo.y = acc[dt][1] * inv;
        lo.z = acc[dt][2] * inv; lo.w = acc[dt][3] * inv;
        hi.x = acc[dt][4] * inv; hi.y = acc[dt][5] * inv;
        hi.z = acc[dt][6] * inv; hi.w = acc[dt][7] * inv;
        *(float4*)o       = lo;
        *(float4*)(o + 4) = hi;
    }
}

// ---------------------------------------------------------------------------
extern "C" void kernel_launch(void* const* d_in, const int* in_sizes, int n_in,
                              void* d_out, int out_size, void* d_ws, size_t ws_size,
                              hipStream_t stream)
{
    const float* q  = (const float*)d_in[0];
    const float* k  = (const float*)d_in[1];
    const float* v  = (const float*)d_in[2];
    const float* wq = (const float*)d_in[3];
    const float* wk = (const float*)d_in[4];
    const float* wv = (const float*)d_in[5];
    float* out = (float*)d_out;

    const size_t elems = (size_t)BB * NN * DD;       // 8,388,608
    _Float16* hq = (_Float16*)d_ws;                  // 16 MB each, 48 MB total
    _Float16* hk = hq + elems;
    _Float16* hv = hk + elems;

    dim3 grid(WW, BB);
    conv_rope_f16_kernel<<<grid, 256, 0, stream>>>(q, k, v, wq, wk, wv, hq, hk, hv);
    local_attn_wmma_kernel<<<grid, 256, 0, stream>>>(hq, hk, hv, out);
}
